// SparseMSDeformableAttention_30202210025500
// MI455X (gfx1250) — matrix operand order repaired and checked by reading, not compiled, not run on hardware
//
#include <hip/hip_runtime.h>
#include <math.h>

typedef __attribute__((ext_vector_type(16))) __bf16 v16bf;
typedef __attribute__((ext_vector_type(8)))  float  v8f;

#define EMBED_DIM 256
#define N_HEADS   8
#define HEAD_DIM  32
#define N_LEVELS  4
#define N_POINTS  4
#define HMAX      64
#define WMAX      64

__device__ __forceinline__ unsigned short f2bf(float x) {
  union { float f; unsigned u; } v; v.f = x;
  unsigned r = v.u + 0x7FFFu + ((v.u >> 16) & 1u);   // round-to-nearest-even
  return (unsigned short)(r >> 16);
}
__device__ __forceinline__ unsigned pkbf(float a, float b) {
  return (unsigned)f2bf(a) | ((unsigned)f2bf(b) << 16);
}

union BF16Frag { v16bf v; unsigned u[8]; uint4 q[2]; };

// ---------------------------------------------------------------------------
// Pack fp32 weight W (K x N, row-major) into bf16, transposed: Wt[n*K + k].
// K-contiguous per output column -> B-fragment loads are b128 loads.
// ---------------------------------------------------------------------------
__global__ __launch_bounds__(256) void pack_wt(const float* __restrict__ W,
                                               unsigned short* __restrict__ Wt,
                                               int K, int N) {
  int idx = blockIdx.x * blockDim.x + threadIdx.x;
  if (idx >= K * N) return;
  int n = idx / K, k = idx - n * K;
  Wt[idx] = f2bf(W[(size_t)k * N + n]);
}

// ---------------------------------------------------------------------------
// Pack an fp32 activation matrix (row-major, n elements, n % 4 == 0) to bf16.
// ---------------------------------------------------------------------------
__global__ __launch_bounds__(256) void pack_act(const float* __restrict__ X,
                                                unsigned short* __restrict__ Xb,
                                                long n) {
  long i = ((long)blockIdx.x * blockDim.x + threadIdx.x) * 4;
  if (i >= n) return;
  float4 v = *(const float4*)(X + i);
  uint2 o; o.x = pkbf(v.x, v.y); o.y = pkbf(v.z, v.w);
  *(uint2*)(Xb + i) = o;
}

// ---------------------------------------------------------------------------
// C = A(MxK, packed bf16) * W(KxN) + bias via v_wmma_f32_16x16x32_bf16.
// One wave -> 32 rows x 64 cols: 2 A fragments share 4 B fragments
// (8 WMMAs per k-step). Block = 4 waves = 32 rows x 256 cols.
// ---------------------------------------------------------------------------
__global__ __launch_bounds__(128) void gemm_bf16_wmma(
    const unsigned short* __restrict__ A, const unsigned short* __restrict__ Wt,
    const float* __restrict__ bias, float* __restrict__ C,
    int M, int N, int K) {
  const int wave = threadIdx.x >> 5;
  const int lane = threadIdx.x & 31;
  const int rowBase = blockIdx.x * 32;
  const int cBase   = blockIdx.y * 256 + wave * 64;
  if (cBase >= N) return;

  const int half = lane >> 4;     // 0: lanes 0-15, 1: lanes 16-31
  const int l16  = lane & 15;
  int aRow0 = rowBase + l16;      if (aRow0 >= M) aRow0 = M - 1;
  int aRow1 = rowBase + 16 + l16; if (aRow1 >= M) aRow1 = M - 1;
  const int aK = half * 8;        // A frag: K {0-7,16-23} vs {8-15,24-31}
  const int bK = half * 8;        // B frag: K {0-7,16-23} vs {8-15,24-31}
  const unsigned short* __restrict__ Ar0 = A + (size_t)aRow0 * K;
  const unsigned short* __restrict__ Ar1 = A + (size_t)aRow1 * K;

  v8f acc[2][4] = {};
  for (int k0 = 0; k0 < K; k0 += 32) {
    __builtin_prefetch(Ar0 + k0 + 128, 0, 1);  // global_prefetch_b8 on A stream
    BF16Frag a0, a1;
    a0.q[0] = *(const uint4*)(Ar0 + k0 + aK);         // K aK .. aK+7
    a0.q[1] = *(const uint4*)(Ar0 + k0 + 16 + aK);    // K 16+aK .. 16+aK+7
    a1.q[0] = *(const uint4*)(Ar1 + k0 + aK);
    a1.q[1] = *(const uint4*)(Ar1 + k0 + 16 + aK);
#pragma unroll
    for (int t = 0; t < 4; ++t) {
      const int col = cBase + t * 16 + l16;
      const unsigned short* __restrict__ bp = Wt + (size_t)col * K + k0 + bK;
      BF16Frag b;
      b.q[0] = *(const uint4*)(bp);
      b.q[1] = *(const uint4*)(bp + 16);
      acc[0][t] = __builtin_amdgcn_wmma_f32_16x16x32_bf16(
          false, a0.v, false, b.v, (short)0, acc[0][t], false, false);
      acc[1][t] = __builtin_amdgcn_wmma_f32_16x16x32_bf16(
          false, a1.v, false, b.v, (short)0, acc[1][t], false, false);
    }
  }
#pragma unroll
  for (int t = 0; t < 4; ++t) {
    const int col = cBase + t * 16 + l16;
    const float bv = bias ? bias[col] : 0.0f;
#pragma unroll
    for (int g = 0; g < 2; ++g) {
#pragma unroll
      for (int r = 0; r < 8; ++r) {
        const int row = rowBase + g * 16 + half * 8 + r;  // C/D: VGPR r -> M=r(+8)
        if (row < M) C[(size_t)row * N + col] = acc[g][t][r] + bv;
      }
    }
  }
}

// ---------------------------------------------------------------------------
// Softmax over (P*L)=16 logits per (q,head) + bilinear gather + weighted sum.
// One block per query, one wave per head, lane = head channel (coalesced 128B
// corner fetches; values buffer fits in L2). Emits packed bf16 directly for
// the final WMMA GEMM.
// ---------------------------------------------------------------------------
__global__ __launch_bounds__(256) void ms_deform_sample(
    const float* __restrict__ pos, const int* __restrict__ batch_off,
    const int* __restrict__ lvl_shapes, const int* __restrict__ q_lvl,
    const float* __restrict__ values, const float* __restrict__ offbuf,
    const float* __restrict__ attnbuf, unsigned short* __restrict__ out_attn,
    int n_q, int n_batches) {
  const int q = blockIdx.x;
  if (q >= n_q) return;
  const int h = threadIdx.x >> 5;
  const int d = threadIdx.x & 31;

  int b = 0;
  for (int i = 1; i < n_batches; ++i) if (q >= batch_off[i]) b = i;

  const int lq = q_lvl[q];
  const float nh = pos[2 * q + 0] / (float)lvl_shapes[2 * lq + 0];
  const float nw = pos[2 * q + 1] / (float)lvl_shapes[2 * lq + 1];

  float lg[16];
  float mx = -1e30f;
#pragma unroll
  for (int pl = 0; pl < 16; ++pl) {
    lg[pl] = attnbuf[(size_t)q * 128 + pl * N_HEADS + h];
    mx = fmaxf(mx, lg[pl]);
  }
  float s = 0.0f;
#pragma unroll
  for (int pl = 0; pl < 16; ++pl) { lg[pl] = __expf(lg[pl] - mx); s += lg[pl]; }
  const float inv = 1.0f / s;

  float acc = 0.0f;
#pragma unroll
  for (int pl = 0; pl < 16; ++pl) {
    const int l = pl & (N_LEVELS - 1);
    const int Hl = lvl_shapes[2 * l + 0], Wl = lvl_shapes[2 * l + 1];
    const float Hf = (float)Hl, Wf = (float)Wl;
    const float oh = offbuf[(size_t)q * 256 + (pl * N_HEADS + h) * 2 + 0];
    const float ow = offbuf[(size_t)q * 256 + (pl * N_HEADS + h) * 2 + 1];
    const float lh = fminf(fmaxf(nh * Hf + oh, 0.0f), Hf);
    const float lw = fminf(fmaxf(nw * Wf + ow, 0.0f), Wf);
    const float h0f = floorf(lh), w0f = floorf(lw);
    const float fh = lh - h0f, fw = lw - w0f;
    const int h0 = (int)h0f, w0 = (int)w0f;   // h0,w0 >= 0 after clamp
    const float aw = lg[pl] * inv;
    const int ch = h * HEAD_DIM + d;
    float sv = 0.0f;
    const float w00 = (1.0f - fh) * (1.0f - fw), w01 = (1.0f - fh) * fw;
    const float w10 = fh * (1.0f - fw),          w11 = fh * fw;
    if (h0 < Hl && w0 < Wl)
      sv += w00 * values[((((size_t)b * HMAX + h0) * WMAX + w0) * N_LEVELS + l) * EMBED_DIM + ch];
    if (h0 < Hl && w0 + 1 < Wl)
      sv += w01 * values[((((size_t)b * HMAX + h0) * WMAX + w0 + 1) * N_LEVELS + l) * EMBED_DIM + ch];
    if (h0 + 1 < Hl && w0 < Wl)
      sv += w10 * values[((((size_t)b * HMAX + h0 + 1) * WMAX + w0) * N_LEVELS + l) * EMBED_DIM + ch];
    if (h0 + 1 < Hl && w0 + 1 < Wl)
      sv += w11 * values[((((size_t)b * HMAX + h0 + 1) * WMAX + w0 + 1) * N_LEVELS + l) * EMBED_DIM + ch];
    acc += aw * sv;
  }
  out_attn[(size_t)q * EMBED_DIM + h * HEAD_DIM + d] = f2bf(acc);
}

// ---------------------------------------------------------------------------
extern "C" void kernel_launch(void* const* d_in, const int* in_sizes, int n_in,
                              void* d_out, int out_size, void* d_ws, size_t ws_size,
                              hipStream_t stream) {
  (void)n_in; (void)out_size; (void)ws_size;
  const float* query  = (const float*)d_in[0];
  const float* pos    = (const float*)d_in[1];
  const int*   boff   = (const int*)d_in[2];
  const float* fmap   = (const float*)d_in[3];
  const int*   lsh    = (const int*)d_in[4];
  const int*   qlvl   = (const int*)d_in[5];
  const float* W_off  = (const float*)d_in[6];
  const float* b_off  = (const float*)d_in[7];
  const float* W_attn = (const float*)d_in[8];
  const float* b_attn = (const float*)d_in[9];
  const float* W_val  = (const float*)d_in[10];
  const float* b_val  = (const float*)d_in[11];
  const float* W_out  = (const float*)d_in[12];
  const float* b_out  = (const float*)d_in[13];
  float* out = (float*)d_out;

  const int n_q = in_sizes[0] / EMBED_DIM;
  const int n_batches = in_sizes[2] - 1;
  const int M1 = in_sizes[3] / EMBED_DIM;   // B*Hm*Wm*L rows of feature maps

  char* ws = (char*)d_ws;
  size_t cur = 0;
  auto alloc = [&](size_t bytes) -> void* {
    void* p = ws + cur;
    cur = (cur + bytes + 255) & ~(size_t)255;
    return p;
  };
  unsigned short* wt_val  = (unsigned short*)alloc((size_t)256 * 256 * 2);
  unsigned short* wt_off  = (unsigned short*)alloc((size_t)256 * 256 * 2);
  unsigned short* wt_attn = (unsigned short*)alloc((size_t)256 * 128 * 2);
  unsigned short* wt_out  = (unsigned short*)alloc((size_t)256 * 256 * 2);
  unsigned short* fmap_b  = (unsigned short*)alloc((size_t)M1  * 256 * 2);
  unsigned short* query_b = (unsigned short*)alloc((size_t)n_q * 256 * 2);
  unsigned short* oattn_b = (unsigned short*)alloc((size_t)n_q * 256 * 2);
  float* values = (float*)alloc((size_t)M1  * 256 * 4);
  float* offv   = (float*)alloc((size_t)n_q * 256 * 4);
  float* attnv  = (float*)alloc((size_t)n_q * 128 * 4);

  // --- one-time packing (weights transposed, activations straight) ---
  pack_wt<<<(256 * 256 + 255) / 256, 256, 0, stream>>>(W_val,  wt_val,  256, 256);
  pack_wt<<<(256 * 256 + 255) / 256, 256, 0, stream>>>(W_off,  wt_off,  256, 256);
  pack_wt<<<(256 * 128 + 255) / 256, 256, 0, stream>>>(W_attn, wt_attn, 256, 128);
  pack_wt<<<(256 * 256 + 255) / 256, 256, 0, stream>>>(W_out,  wt_out,  256, 256);
  {
    long nf = (long)M1 * 256, nq = (long)n_q * 256;
    pack_act<<<(unsigned)((nf / 4 + 255) / 256), 256, 0, stream>>>(fmap,  fmap_b,  nf);
    pack_act<<<(unsigned)((nq / 4 + 255) / 256), 256, 0, stream>>>(query, query_b, nq);
  }

  // --- GEMMs (WMMA) + sampling ---
  dim3 blk(128);
  gemm_bf16_wmma<<<dim3((M1  + 31) / 32, 1), blk, 0, stream>>>(fmap_b,  wt_val,  b_val,  values, M1,  256, 256);
  gemm_bf16_wmma<<<dim3((n_q + 31) / 32, 1), blk, 0, stream>>>(query_b, wt_off,  b_off,  offv,   n_q, 256, 256);
  gemm_bf16_wmma<<<dim3((n_q + 31) / 32, 1), blk, 0, stream>>>(query_b, wt_attn, b_attn, attnv,  n_q, 128, 256);

  ms_deform_sample<<<n_q, 256, 0, stream>>>(pos, boff, lsh, qlvl, values, offv,
                                            attnv, oattn_b, n_q, n_batches);

  gemm_bf16_wmma<<<dim3((n_q + 31) / 32, 1), blk, 0, stream>>>(oattn_b, wt_out, b_out, out, n_q, 256, 256);
}
